// BitLinear_1340029796975
// MI455X (gfx1250) — compile-verified, matrix operand order checked by reading, not run on hardware
//
#include <hip/hip_runtime.h>
#include <hip/hip_bf16.h>

#define IN_F   2048
#define OUT_F  8192
#define TOKENS 8192            // 4 * 2048
#define KB     (IN_F / 64)     // 32 k-blocks of 64
#define EPS_RMS 1e-5f

typedef __attribute__((ext_vector_type(8))) int v8i;

// Byte offset inside a 1024B WMMA iu8 fragment (16 rows x 64 k), following the
// ISA 8-bit A-matrix layout: lane = row + 16*h, VGPR v = 2*q + p, byte b,
// where k = 16*q + 8*h + 4*p + b.
__device__ __forceinline__ unsigned frag_byte_off(unsigned row, unsigned kr) {
    unsigned q = kr >> 4;
    unsigned h = (kr >> 3) & 1u;
    unsigned p = (kr >> 2) & 1u;
    unsigned b = kr & 3u;
    unsigned lane = row + (h << 4);
    unsigned v = (q << 1) | p;
    return lane * 32u + v * 4u + b;
}

// ---------------- weight global reductions: e = mean(w), s = mean(|w|) -------
__global__ void wsum_partial(const float* __restrict__ w, float* __restrict__ part) {
    __shared__ float sS[256];
    __shared__ float sA[256];
    int tid = threadIdx.x;
    size_t idx = (size_t)blockIdx.x * 256 + tid;
    const size_t total = (size_t)OUT_F * IN_F;
    float s = 0.f, a = 0.f;
    for (size_t i = idx; i < total; i += 256u * 256u) {
        float v = w[i];
        s += v;
        a += fabsf(v);
    }
    sS[tid] = s; sA[tid] = a;
    __syncthreads();
    for (int off = 128; off > 0; off >>= 1) {
        if (tid < off) { sS[tid] += sS[tid + off]; sA[tid] += sA[tid + off]; }
        __syncthreads();
    }
    if (tid == 0) { part[blockIdx.x] = sS[0]; part[256 + blockIdx.x] = sA[0]; }
}

__global__ void wsum_final(const float* __restrict__ part, float* __restrict__ scal) {
    __shared__ float sS[256];
    __shared__ float sA[256];
    int tid = threadIdx.x;
    sS[tid] = part[tid]; sA[tid] = part[256 + tid];
    __syncthreads();
    for (int off = 128; off > 0; off >>= 1) {
        if (tid < off) { sS[tid] += sS[tid + off]; sA[tid] += sA[tid + off]; }
        __syncthreads();
    }
    if (tid == 0) {
        const float inv = 1.0f / (float)((size_t)OUT_F * IN_F);
        scal[0] = sS[0] * inv;                  // e = mean(w)
        scal[1] = fmaxf(sA[0] * inv, 1e-8f);    // s = max(mean|w|, 1e-8)
    }
}

// ---------------- weight ternary quant -> packed B fragments -----------------
__global__ void wquant(const float* __restrict__ w, const float* __restrict__ scal,
                       char* __restrict__ Bp) {
    const float e = scal[0];
    unsigned flat = (blockIdx.x * 256u + threadIdx.x) * 8u;  // row-major (n, k)
    unsigned n  = flat >> 11;        // / IN_F
    unsigned k0 = flat & 2047u;
    unsigned nTile = n >> 4, row = n & 15u;
#pragma unroll
    for (int j = 0; j < 8; ++j) {
        unsigned k = k0 + j;
        float v = w[flat + j];
        char sgn = (v > e) ? (char)1 : ((v < e) ? (char)-1 : (char)0);
        unsigned kb = k >> 6, kr = k & 63u;
        size_t off = ((size_t)(nTile * KB + kb) << 10) + frag_byte_off(row, kr);
        Bp[off] = sgn;
    }
}

// ---------------- rmsnorm + activation int8 quant -> packed A fragments ------
__global__ void xquant(const float* __restrict__ x, const float* __restrict__ g,
                       char* __restrict__ Ap, float* __restrict__ invScale) {
    __shared__ float red[256];
    int tid = threadIdx.x;
    int m = blockIdx.x;                      // token
    const float* xr = x + (size_t)m * IN_F;
    int k0 = tid * 8;

    float v[8];
    float ss = 0.f;
#pragma unroll
    for (int j = 0; j < 8; ++j) {
        float t = xr[k0 + j];
        t = fminf(fmaxf(t, -100.f), 100.f);  // clip
        v[j] = t;
        ss += t * t;
    }
    red[tid] = ss;
    __syncthreads();
    for (int off = 128; off > 0; off >>= 1) {
        if (tid < off) red[tid] += red[tid + off];
        __syncthreads();
    }
    float var = red[0] * (1.0f / IN_F);
    var = fmaxf(var, EPS_RMS);
    float r = rsqrtf(var + EPS_RMS);
    __syncthreads();

    float mx = 0.f;
#pragma unroll
    for (int j = 0; j < 8; ++j) {
        float t = v[j] * r;
        t = fminf(fmaxf(t, -10.f), 10.f) * g[k0 + j];  // rmsnorm output
        t = fminf(fmaxf(t, -50.f), 50.f);              // activation_quant clip
        v[j] = t;
        mx = fmaxf(mx, fabsf(t));
    }
    red[tid] = mx;
    __syncthreads();
    for (int off = 128; off > 0; off >>= 1) {
        if (tid < off) red[tid] = fmaxf(red[tid], red[tid + off]);
        __syncthreads();
    }
    float maxv  = fmaxf(red[0], 1e-4f);
    float scale = 127.f / maxv;
    if (tid == 0) invScale[m] = maxv * (1.f / 127.f);

    unsigned mTile = (unsigned)m >> 4, row = (unsigned)m & 15u;
#pragma unroll
    for (int j = 0; j < 8; ++j) {
        unsigned k = (unsigned)(k0 + j);
        float q = rintf(v[j] * scale);               // ties-to-even == jnp.round
        q = fminf(fmaxf(q, -128.f), 127.f);
        unsigned kb = k >> 6, kr = k & 63u;
        size_t off = ((size_t)(mTile * KB + kb) << 10) + frag_byte_off(row, kr);
        Ap[off] = (char)(int)q;
    }
}

// ---------------- int8 x ternary GEMM via V_WMMA_I32_16X16X64_IU8 ------------
// 256 threads = 8 wave32. Block tile: 128(M) x 128(N). Wave tile: 64 x 32
// = 4x2 fragments of 16x16; K loop over 32 blocks of 64.
__global__ void __launch_bounds__(256) bitlinear_gemm(
    const char* __restrict__ Ap, const char* __restrict__ Bp,
    const float* __restrict__ invScale, const float* __restrict__ scal,
    float* __restrict__ y) {
    const float wscale = scal[1];
    int tid  = threadIdx.x;
    int lane = tid & 31;
    int w    = tid >> 5;         // wave 0..7
    int wm   = w >> 2;           // 0..1
    int wn   = w & 3;            // 0..3
    int mT0  = blockIdx.y * 8 + wm * 4;   // 4 consecutive M tiles
    int nT0  = blockIdx.x * 8 + wn * 2;   // 2 consecutive N tiles

    const v8i* aBase[4];
    const v8i* bBase[2];
#pragma unroll
    for (int i = 0; i < 4; ++i)
        aBase[i] = (const v8i*)(Ap + (((size_t)(mT0 + i) * KB) << 10) + lane * 32);
#pragma unroll
    for (int j = 0; j < 2; ++j)
        bBase[j] = (const v8i*)(Bp + (((size_t)(nT0 + j) * KB) << 10) + lane * 32);

    v8i acc[4][2];
#pragma unroll
    for (int i = 0; i < 4; ++i)
#pragma unroll
        for (int j = 0; j < 2; ++j)
            acc[i][j] = (v8i){0, 0, 0, 0, 0, 0, 0, 0};

    for (int kb = 0; kb < KB; ++kb) {
        v8i a[4], b[2];
#pragma unroll
        for (int i = 0; i < 4; ++i) a[i] = aBase[i][kb * 32];  // 1024B fragment stride
#pragma unroll
        for (int j = 0; j < 2; ++j) b[j] = bBase[j][kb * 32];
        if (kb + 1 < KB) {  // L2 prefetch of next-k A fragment (global_prefetch_b8)
            __builtin_prefetch((const char*)aBase[0] + (size_t)(kb + 1) * 1024, 0, 1);
        }
#pragma unroll
        for (int i = 0; i < 4; ++i)
#pragma unroll
            for (int j = 0; j < 2; ++j)
                acc[i][j] = __builtin_amdgcn_wmma_i32_16x16x64_iu8(
                    /*sgn_a=*/true, a[i], /*sgn_b=*/true, b[j], acc[i][j],
                    /*reuse_a=*/false, /*reuse_b=*/false);
    }

    // Epilogue: C layout — lane 0..15: M rows 0..7 in VGPR0..7, lane 16..31: rows 8..15.
    int col  = lane & 15;
    int half = lane >> 4;
#pragma unroll
    for (int i = 0; i < 4; ++i) {
        int mBase = (mT0 + i) * 16 + half * 8;
#pragma unroll
        for (int j = 0; j < 2; ++j) {
            int n = (nT0 + j) * 16 + col;
#pragma unroll
            for (int v = 0; v < 8; ++v) {
                int m = mBase + v;
                y[(size_t)m * OUT_F + n] = (float)acc[i][j][v] * wscale * invScale[m];
            }
        }
    }
}

extern "C" void kernel_launch(void* const* d_in, const int* in_sizes, int n_in,
                              void* d_out, int out_size, void* d_ws, size_t ws_size,
                              hipStream_t stream) {
    const float* x      = (const float*)d_in[0];   // (4,2048,2048) f32
    const float* weight = (const float*)d_in[1];   // (8192,2048)  f32
    const float* g      = (const float*)d_in[2];   // (2048,)      f32
    float*       y      = (float*)d_out;           // (4,2048,8192) f32

    char* ws = (char*)d_ws;
    char*  Bp       = ws;                                   // 16 MB packed B signs
    char*  Ap       = ws + ((size_t)16 << 20);              // 16 MB packed A int8
    float* invScale = (float*)(ws + ((size_t)32 << 20));    // 32 KB per-token scale
    float* part     = (float*)(ws + ((size_t)32 << 20) + 32768);          // 2 KB
    float* scal     = (float*)(ws + ((size_t)32 << 20) + 32768 + 2048);   // e, s

    wsum_partial<<<256, 256, 0, stream>>>(weight, part);
    wsum_final  <<<1,   256, 0, stream>>>(part, scal);
    wquant      <<<(OUT_F * IN_F) / (256 * 8), 256, 0, stream>>>(weight, scal, Bp);
    xquant      <<<TOKENS, 256, 0, stream>>>(x, g, Ap, invScale);

    dim3 grid(OUT_F / 128, TOKENS / 128);   // 64 x 64 blocks
    bitlinear_gemm<<<grid, 256, 0, stream>>>(Ap, Bp, invScale, scal, y);
}
